// TransformerBlock_2534030705250
// MI455X (gfx1250) — matrix operand order repaired and checked by reading, not compiled, not run on hardware
//
#include <hip/hip_runtime.h>
#include <hip/hip_bf16.h>

// ---------------------------------------------------------------------------
// Transformer block for MI455X (gfx1250, wave32, WMMA bf16 w/ f32 accumulate)
//   EMB=1024, HEADS=16, HEAD_DIM=64, B=2, S=2048, DFF=4096, tokens T=4096
//   GEMM A-tiles staged via Tensor Data Mover (tensor_load_to_lds, TENSORcnt)
// ---------------------------------------------------------------------------

#define EMB      1024
#define HEADS    16
#define HEAD_DIM 64
#define SEQ      2048
#define BATCH    2
#define TOK      (BATCH * SEQ)          // 4096
#define DFF      (4 * EMB)              // 4096

typedef __bf16        v16bf __attribute__((ext_vector_type(16)));
typedef __bf16        v8bf  __attribute__((ext_vector_type(8)));
typedef float         v8f   __attribute__((ext_vector_type(8)));
typedef unsigned int  u32x4 __attribute__((ext_vector_type(4)));
typedef int           i32x8 __attribute__((ext_vector_type(8)));
typedef int           i32x4 __attribute__((ext_vector_type(4)));

static __device__ inline v8f wmma_bf16(v16bf a, v16bf b, v8f c) {
    // D = A(16x32 bf16) * B(32x16 bf16) + C(16x16 f32)
    return __builtin_amdgcn_wmma_f32_16x16x32_bf16(
        false, a, false, b, (short)0, c, false, false);
}

static __device__ inline v16bf comb16(v8bf lo, v8bf hi) {
    v16bf r;
#pragma unroll
    for (int i = 0; i < 8; ++i) { r[i] = lo[i]; r[i + 8] = hi[i]; }
    return r;
}

static __device__ inline v8f zero8() {
    v8f z = {0.f, 0.f, 0.f, 0.f, 0.f, 0.f, 0.f, 0.f};
    return z;
}

// Issue one TDM 2-D tile load: tile_dim0 x tile_dim1 elements of 2 bytes,
// row stride = `stride` elements in global memory, padded in LDS by 32B per
// 64B row (pad_interval=3 -> every 16 DWORDs, pad_amount=7 -> 8 DWORDs).
// Resulting LDS row stride = 96 bytes (48 bf16).
static __device__ inline void tdm_load_tile_bf16(const __bf16* gsrc,
                                                 unsigned lds_off, int strideElems) {
    const unsigned long long ga = (unsigned long long)(const void*)gsrc;
    u32x4 g0;
    g0[0] = 1u;                                            // count=1 (valid D#)
    g0[1] = lds_off;                                       // lds_addr
    g0[2] = (unsigned)ga;                                  // global_addr[31:0]
    g0[3] = ((unsigned)(ga >> 32) & 0x01FFFFFFu) | (2u << 30);  // addr[56:32], type=2
    const unsigned TDIM = 1u << 20;                        // generous tensor dims (no OOB)
    i32x8 g1;
    g1[0] = (1 << 16) | (1 << 20) | (3 << 22) | (7 << 25); // data_size=2B, pad en/int/amt
    g1[1] = (int)((TDIM & 0xFFFFu) << 16);                 // tensor_dim0 lo16
    g1[2] = (int)((TDIM >> 16) | ((TDIM & 0xFFFFu) << 16));// dim0 hi16 | dim1 lo16
    g1[3] = (int)((TDIM >> 16) | (32u << 16));             // dim1 hi16 | tile_dim0=32
    g1[4] = 128;                                           // tile_dim1=128, tile_dim2=0
    g1[5] = strideElems;                                   // tensor_dim0_stride[31:0]
    g1[6] = 0;                                             // stride0 hi, stride1 lo
    g1[7] = 0;
    i32x4 g2 = {0, 0, 0, 0};
    i32x4 g3 = {0, 0, 0, 0};
    asm volatile("tensor_load_to_lds %0, %1, %2, %3"
                 :: "s"(g0), "s"(g1), "s"(g2), "s"(g3) : "memory");
}

// ---------------------------------------------------------------------------
// fp32 -> bf16 cast (weights)
// ---------------------------------------------------------------------------
__global__ __launch_bounds__(256)
void cast_bf16_kernel(const float* __restrict__ in, __bf16* __restrict__ out, int n) {
    int i = blockIdx.x * 256 + threadIdx.x;
    if (i < n) out[i] = (__bf16)in[i];
}

// ---------------------------------------------------------------------------
// LayerNorm over D=1024, one block (256 threads) per row, bf16 output
// ---------------------------------------------------------------------------
__global__ __launch_bounds__(256)
void layernorm_bf16_kernel(const float* __restrict__ x,
                           const float* __restrict__ gamma,
                           const float* __restrict__ beta,
                           __bf16* __restrict__ out) {
    __shared__ float red0[8], red1[8];
    const int row  = blockIdx.x;
    const int tid  = threadIdx.x;
    const int lane = tid & 31, wave = tid >> 5;

    const float4 v = ((const float4*)(x + (size_t)row * EMB))[tid];
    float s  = v.x + v.y + v.z + v.w;
    float ss = v.x * v.x + v.y * v.y + v.z * v.z + v.w * v.w;
#pragma unroll
    for (int off = 16; off >= 1; off >>= 1) {
        s  += __shfl_xor(s,  off);
        ss += __shfl_xor(ss, off);
    }
    if (lane == 0) { red0[wave] = s; red1[wave] = ss; }
    __syncthreads();
    float ts = 0.f, tss = 0.f;
#pragma unroll
    for (int w = 0; w < 8; ++w) { ts += red0[w]; tss += red1[w]; }
    const float mean = ts * (1.0f / EMB);
    const float var  = tss * (1.0f / EMB) - mean * mean;
    const float rstd = rsqrtf(var + 1e-5f);

    const int c = tid * 4;
    const float4 g = ((const float4*)gamma)[tid];
    const float4 b = ((const float4*)beta)[tid];
    __bf16* o = out + (size_t)row * EMB + c;
    o[0] = (__bf16)(g.x * (v.x - mean) * rstd + b.x);
    o[1] = (__bf16)(g.y * (v.y - mean) * rstd + b.y);
    o[2] = (__bf16)(g.z * (v.z - mean) * rstd + b.z);
    o[3] = (__bf16)(g.w * (v.w - mean) * rstd + b.w);
}

// ---------------------------------------------------------------------------
// Tiled bf16 GEMM: C[M,N] = A[M,K] @ B[K,N]  (row-major), f32 accumulate.
//   Block = 256 threads (8 waves), tile 128x128, K-step 32.
//   A tile (128x32) staged by the Tensor Data Mover into LDS (96B row stride).
//   B tile staged transposed by the threads (TDM cannot transpose).
//   Wave grid 4(m) x 2(n); each wave: 32x64 = 2x4 WMMA 16x16 accumulators.
// Epilogue modes:
//   0: QKV scatter  -> outB[((b*H+h)*S + s)*64 + hd]  (bf16)
//   1: f32 out = acc + bias[n] + resid[m*N+n]
//   2: bf16 out = gelu_tanh(acc + bias[n])
// ---------------------------------------------------------------------------
#define GM_QKV  0
#define GM_F32  1
#define GM_GELU 2

#define AS_STRIDE 48   // bf16 elems per LDS row after TDM padding (96 bytes)

template <int MODE>
__global__ __launch_bounds__(256)
void gemm_bf16_kernel(const __bf16* __restrict__ A, const __bf16* __restrict__ Bm,
                      const float* __restrict__ bias, const float* __restrict__ resid,
                      float* __restrict__ outF, __bf16* __restrict__ outB,
                      int M, int N, int K) {
    __shared__ __align__(16) __bf16 As[128 * AS_STRIDE]; // TDM dest, 12 KB
    __shared__ __align__(16) __bf16 Bt[128][40];         // B transposed: Bt[n][k]

    const int tid   = threadIdx.x;
    const int lane  = tid & 31;
    const int wave  = tid >> 5;
    const int waveM = wave >> 1;       // 0..3
    const int waveN = wave & 1;        // 0..1
    const int l15   = lane & 15;
    const int lh    = lane >> 4;       // 0/1
    const int bm    = blockIdx.y * 128;
    const int bn    = blockIdx.x * 128;

    const unsigned asBase = (unsigned)(size_t)&As[0];    // LDS byte offset

    v8f acc[2][4];
#pragma unroll
    for (int mi = 0; mi < 2; ++mi)
#pragma unroll
        for (int ni = 0; ni < 4; ++ni) acc[mi][ni] = zero8();

    for (int k0 = 0; k0 < K; k0 += 32) {
        // --- TDM: wave 0 DMAs the whole 128x32 A tile into LDS ---
        if (wave == 0)
            tdm_load_tile_bf16(A + (size_t)bm * K + k0, asBase, K);

        // --- stage B tile 32x128 transposed into Bt[n][k] (overlaps TDM) ---
        {
            const int kr = tid >> 3;          // 0..31
            const int nc = (tid & 7) * 16;    // 0..112
            const v8bf* src = (const v8bf*)(Bm + (size_t)(k0 + kr) * N + bn + nc);
            v8bf b0 = src[0], b1 = src[1];
#pragma unroll
            for (int i = 0; i < 8; ++i) {
                Bt[nc + i][kr]     = b0[i];
                Bt[nc + 8 + i][kr] = b1[i];
            }
            if (k0 + 32 < K)
                __builtin_prefetch(Bm + (size_t)(k0 + 32 + kr) * N + bn + nc, 0, 1);
        }

        if (wave == 0) __builtin_amdgcn_s_wait_tensorcnt(0);
        __syncthreads();

        // --- fragments ---
        v16bf afrag[2], bfrag[4];
#pragma unroll
        for (int mi = 0; mi < 2; ++mi) {
            const int row = waveM * 32 + mi * 16 + l15;
            const __bf16* p = &As[row * AS_STRIDE + lh * 8];
            afrag[mi] = comb16(*(const v8bf*)p, *(const v8bf*)(p + 16));
        }
#pragma unroll
        for (int ni = 0; ni < 4; ++ni) {
            const int col = waveN * 64 + ni * 16 + l15;
            const __bf16* p = &Bt[col][lh * 8];
            bfrag[ni] = comb16(*(const v8bf*)p, *(const v8bf*)(p + 16));
        }
#pragma unroll
        for (int mi = 0; mi < 2; ++mi)
#pragma unroll
            for (int ni = 0; ni < 4; ++ni)
                acc[mi][ni] = wmma_bf16(afrag[mi], bfrag[ni], acc[mi][ni]);
        __syncthreads();
    }

    // --- epilogue ---
#pragma unroll
    for (int mi = 0; mi < 2; ++mi) {
#pragma unroll
        for (int ni = 0; ni < 4; ++ni) {
#pragma unroll
            for (int r = 0; r < 8; ++r) {
                const int m = bm + waveM * 32 + mi * 16 + lh * 8 + r;
                const int n = bn + waveN * 64 + ni * 16 + l15;
                const float v = acc[mi][ni][r];
                if (MODE == GM_QKV) {
                    const int b = m >> 11, srow = m & (SEQ - 1);
                    const int h = n >> 6,  hd  = n & 63;
                    outB[(((size_t)(b * HEADS + h)) * SEQ + srow) * HEAD_DIM + hd] = (__bf16)v;
                } else if (MODE == GM_F32) {
                    outF[(size_t)m * N + n] = v + bias[n] + resid[(size_t)m * N + n];
                } else { // GM_GELU
                    const float u = v + bias[n];
                    const float g = 0.5f * u *
                        (1.0f + tanhf(0.7978845608028654f * (u + 0.044715f * u * u * u)));
                    outB[(size_t)m * N + n] = (__bf16)g;
                }
            }
        }
    }
}

// ---------------------------------------------------------------------------
// Flash attention (causal). q/k/v in bf16 [B*H, S, 64]. ctx out bf16 [T, EMB]
// with col = h*64 + hd  (matches einsum "bhqk,bhkd->bqhd" reshape).
// Block 256 threads = 8 waves; each wave owns 16 queries; 32 keys per step.
// ---------------------------------------------------------------------------
__global__ __launch_bounds__(256)
void attn_kernel(const __bf16* __restrict__ q, const __bf16* __restrict__ k,
                 const __bf16* __restrict__ v, __bf16* __restrict__ ctx) {
    __shared__ __align__(16) __bf16 Psh[8][16][32];  // per-wave P tile

    const int tid  = threadIdx.x;
    const int lane = tid & 31;
    const int wave = tid >> 5;
    const int l15  = lane & 15;
    const int lh   = lane >> 4;
    const int lh8  = lh * 8;

    const int bh = blockIdx.y;             // 0..31
    const int b  = bh >> 4;
    const int h  = bh & 15;
    const int q0 = blockIdx.x * 128 + wave * 16;

    const __bf16* qb = q + (size_t)bh * SEQ * HEAD_DIM;
    const __bf16* kb = k + (size_t)bh * SEQ * HEAD_DIM;
    const __bf16* vb = v + (size_t)bh * SEQ * HEAD_DIM;

    // Q fragments: A-layout, K-dim = head_dim (two 16x32 slabs)
    v16bf qf[2];
#pragma unroll
    for (int kk = 0; kk < 2; ++kk) {
        const __bf16* p = qb + (size_t)(q0 + l15) * HEAD_DIM + kk * 32 + lh * 16;
        qf[kk] = comb16(*(const v8bf*)p, *(const v8bf*)(p + 8));
    }

    float mrun[8], lrun[8];
    v8f o[4];
#pragma unroll
    for (int r = 0; r < 8; ++r) { mrun[r] = -1e30f; lrun[r] = 0.f; }
#pragma unroll
    for (int ni = 0; ni < 4; ++ni) o[ni] = zero8();

    const int nsteps = (q0 + 15) / 32 + 1;
    const float SCALE = 0.125f;            // 1/sqrt(64)

    for (int st = 0; st < nsteps; ++st) {
        const int keyb = st * 32;

        // ---- scores: two 16-key tiles, 2 WMMAs each ----
        v8f s0 = zero8(), s1 = zero8();
        {
            const __bf16* p0 = kb + (size_t)(keyb + l15) * HEAD_DIM + lh * 16;
            v16bf kf;
            kf = *(const v16bf*)p0;                 s0 = wmma_bf16(qf[0], kf, s0);
            kf = *(const v16bf*)(p0 + 32);          s0 = wmma_bf16(qf[1], kf, s0);
            const __bf16* p1 = kb + (size_t)(keyb + 16 + l15) * HEAD_DIM + lh * 16;
            kf = *(const v16bf*)p1;                 s1 = wmma_bf16(qf[0], kf, s1);
            kf = *(const v16bf*)(p1 + 32);          s1 = wmma_bf16(qf[1], kf, s1);
        }

        // ---- streaming softmax (rows live across 16-lane groups) ----
#pragma unroll
        for (int r = 0; r < 8; ++r) {
            const int mrow = q0 + lh8 + r;
            float a0 = s0[r] * SCALE;
            float a1 = s1[r] * SCALE;
            if (keyb + l15      > mrow) a0 = -INFINITY;
            if (keyb + 16 + l15 > mrow) a1 = -INFINITY;

            float rm = fmaxf(a0, a1);
#pragma unroll
            for (int off = 1; off < 16; off <<= 1) rm = fmaxf(rm, __shfl_xor(rm, off));
            const float nm    = fmaxf(mrun[r], rm);
            const float alpha = __expf(mrun[r] - nm);
            const float p0    = __expf(a0 - nm);
            const float p1    = __expf(a1 - nm);
            float rs = p0 + p1;
#pragma unroll
            for (int off = 1; off < 16; off <<= 1) rs += __shfl_xor(rs, off);
            lrun[r] = lrun[r] * alpha + rs;
            mrun[r] = nm;
#pragma unroll
            for (int ni = 0; ni < 4; ++ni) o[ni][r] *= alpha;

            Psh[wave][lh8 + r][l15]      = (__bf16)p0;
            Psh[wave][lh8 + r][16 + l15] = (__bf16)p1;
        }

        // intra-wave LDS exchange: drain DS ops, fence scheduling
        __builtin_amdgcn_wave_barrier();
        asm volatile("s_wait_dscnt 0" ::: "memory");
        __builtin_amdgcn_wave_barrier();

        // ---- P fragment (A-layout 16x32) ----
        const __bf16* pp = &Psh[wave][l15][lh * 16];
        const v16bf pf = comb16(*(const v8bf*)pp, *(const v8bf*)(pp + 8));

        // ---- P @ V : 4 WMMAs over hd tiles of 16 ----
#pragma unroll
        for (int ni = 0; ni < 4; ++ni) {
            v16bf vf;
#pragma unroll
            for (int j = 0; j < 16; ++j)
                vf[j] = vb[(size_t)(keyb + lh * 16 + j) * HEAD_DIM + ni * 16 + l15];
            o[ni] = wmma_bf16(pf, vf, o[ni]);
        }
        __builtin_amdgcn_wave_barrier();
    }

    // ---- normalize + write ctx ----
#pragma unroll
    for (int ni = 0; ni < 4; ++ni) {
#pragma unroll
        for (int r = 0; r < 8; ++r) {
            const int mrow = q0 + lh8 + r;
            const float val = o[ni][r] / lrun[r];
            ctx[(size_t)(b * SEQ + mrow) * EMB + h * HEAD_DIM + ni * 16 + l15] = (__bf16)val;
        }
    }
}

// ---------------------------------------------------------------------------
// Host-side orchestration
// ---------------------------------------------------------------------------
extern "C" void kernel_launch(void* const* d_in, const int* in_sizes, int n_in,
                              void* d_out, int out_size, void* d_ws, size_t ws_size,
                              hipStream_t stream) {
    (void)in_sizes; (void)n_in; (void)out_size; (void)ws_size;

    const float* x  = (const float*)d_in[0];
    const float* Wq = (const float*)d_in[1];
    const float* Wk = (const float*)d_in[2];
    const float* Wv = (const float*)d_in[3];
    const float* Wo = (const float*)d_in[4];
    const float* bo = (const float*)d_in[5];
    const float* W1 = (const float*)d_in[6];
    const float* b1 = (const float*)d_in[7];
    const float* W2 = (const float*)d_in[8];
    const float* b2 = (const float*)d_in[9];
    const float* g1 = (const float*)d_in[10];
    const float* s1 = (const float*)d_in[11];
    const float* g2 = (const float*)d_in[12];
    const float* s2 = (const float*)d_in[13];
    float* out = (float*)d_out;

    char* ws = (char*)d_ws;
    const size_t MB = 1ull << 20;
    __bf16* WqB   = (__bf16*)(ws + 0 * MB);    // 2 MB
    __bf16* WkB   = (__bf16*)(ws + 2 * MB);    // 2 MB
    __bf16* WvB   = (__bf16*)(ws + 4 * MB);    // 2 MB
    __bf16* WoB   = (__bf16*)(ws + 6 * MB);    // 2 MB
    __bf16* W1B   = (__bf16*)(ws + 8 * MB);    // 8 MB
    __bf16* W2B   = (__bf16*)(ws + 16 * MB);   // 8 MB
    __bf16* ln1B  = (__bf16*)(ws + 24 * MB);   // 8 MB
    __bf16* Qb    = (__bf16*)(ws + 32 * MB);   // 8 MB
    __bf16* Kb    = (__bf16*)(ws + 40 * MB);   // 8 MB
    __bf16* Vb    = (__bf16*)(ws + 48 * MB);   // 8 MB
    __bf16* ctxB  = (__bf16*)(ws + 56 * MB);   // 8 MB
    float*  hF    = (float*) (ws + 64 * MB);   // 16 MB
    __bf16* ln2B  = (__bf16*)(ws + 80 * MB);   // 8 MB
    __bf16* ffn1B = (__bf16*)(ws + 88 * MB);   // 32 MB -> total 120 MB

    // 1) weight casts to bf16
    const int nW = EMB * EMB;           // 1M
    const int nF = EMB * DFF;           // 4M
    cast_bf16_kernel<<<nW / 256, 256, 0, stream>>>(Wq, WqB, nW);
    cast_bf16_kernel<<<nW / 256, 256, 0, stream>>>(Wk, WkB, nW);
    cast_bf16_kernel<<<nW / 256, 256, 0, stream>>>(Wv, WvB, nW);
    cast_bf16_kernel<<<nW / 256, 256, 0, stream>>>(Wo, WoB, nW);
    cast_bf16_kernel<<<nF / 256, 256, 0, stream>>>(W1, W1B, nF);
    cast_bf16_kernel<<<nF / 256, 256, 0, stream>>>(W2, W2B, nF);

    // 2) ln1 = LN(x)
    layernorm_bf16_kernel<<<TOK, 256, 0, stream>>>(x, g1, s1, ln1B);

    // 3) Q, K, V = ln1 @ W{q,k,v}, scattered to [B*H, S, 64] bf16
    dim3 gQKV(EMB / 128, TOK / 128);
    gemm_bf16_kernel<GM_QKV><<<gQKV, 256, 0, stream>>>(
        ln1B, WqB, nullptr, nullptr, nullptr, Qb, TOK, EMB, EMB);
    gemm_bf16_kernel<GM_QKV><<<gQKV, 256, 0, stream>>>(
        ln1B, WkB, nullptr, nullptr, nullptr, Kb, TOK, EMB, EMB);
    gemm_bf16_kernel<GM_QKV><<<gQKV, 256, 0, stream>>>(
        ln1B, WvB, nullptr, nullptr, nullptr, Vb, TOK, EMB, EMB);

    // 4) causal flash attention -> ctx
    attn_kernel<<<dim3(SEQ / 128, BATCH * HEADS), 256, 0, stream>>>(Qb, Kb, Vb, ctxB);

    // 5) h = x + ctx @ Wo + bo   (f32)
    gemm_bf16_kernel<GM_F32><<<gQKV, 256, 0, stream>>>(
        ctxB, WoB, bo, x, hF, nullptr, TOK, EMB, EMB);

    // 6) ln2 = LN(h)
    layernorm_bf16_kernel<<<TOK, 256, 0, stream>>>(hF, g2, s2, ln2B);

    // 7) ffn1 = gelu(ln2 @ W1 + b1)  (bf16)
    gemm_bf16_kernel<GM_GELU><<<dim3(DFF / 128, TOK / 128), 256, 0, stream>>>(
        ln2B, W1B, b1, nullptr, nullptr, ffn1B, TOK, DFF, EMB);

    // 8) out = h + ffn1 @ W2 + b2   (f32)
    gemm_bf16_kernel<GM_F32><<<gQKV, 256, 0, stream>>>(
        ffn1B, W2B, b2, hF, out, nullptr, TOK, EMB, DFF);
}
